// LocalSparseAttention_3496103379663
// MI455X (gfx1250) — matrix operand order repaired and checked by reading, not compiled, not run on hardware
//
#include <hip/hip_runtime.h>

typedef __attribute__((ext_vector_type(16))) _Float16 v16h;
typedef __attribute__((ext_vector_type(8)))  _Float16 v8h;
typedef __attribute__((ext_vector_type(2)))  _Float16 v2h;
typedef __attribute__((ext_vector_type(8)))  float    v8f;

#define L_SEQ   4096
#define DHEAD   64
#define WIN     128
#define QTILE   128             // query rows per workgroup (8 waves)
#define NWAVES  8
#define WROWS   16              // query rows per wave
#define CHUNK   64              // keys per KV chunk
#define NCHUNK  ((QTILE + 2*WIN)/CHUNK)   // 6
#define KSTR    72              // f16 row stride sK  (144B, 16B-aligned)
#define VSTR    72              // f16 row stride sVt (144B, 16B-aligned)
#define PSTR    72              // f16 row stride sP
#define NEG_BIG (-1.0e30f)

static __device__ __forceinline__ v16h cat16(v8h a, v8h b) {
  v16h r;
  #pragma unroll
  for (int i = 0; i < 8; ++i) { r[i] = a[i]; r[i + 8] = b[i]; }
  return r;
}

__global__ __launch_bounds__(256)
void lsa_fwd(const float* __restrict__ Qg, const float* __restrict__ Kg,
             const float* __restrict__ Vg, float* __restrict__ Og)
{
  __shared__ __align__(16) _Float16 sK [CHUNK][KSTR];        // K chunk row-major [key][dim]
  __shared__ __align__(16) _Float16 sVt[DHEAD][VSTR];        // V chunk transposed [dim][key]
  __shared__ __align__(16) _Float16 sP [NWAVES][WROWS][PSTR];// per-wave P tile [row][key]

  const int tid  = threadIdx.x;
  const int lane = tid & 31;
  const int wave = tid >> 5;        // 0..7
  const int hl   = lane >> 4;       // half-wave 0/1
  const int l16  = lane & 15;

  const int q0 = blockIdx.x * QTILE;
  const long long base = (long long)blockIdx.y * L_SEQ * DHEAD;
  const float* Q = Qg + base;
  const float* K = Kg + base;
  const float* V = Vg + base;
  float*       O = Og + base;

  // ---- Q A-fragments, 1/sqrt(D) folded into the f16 conversion ----
  const int qrow = q0 + wave * WROWS + l16;
  v16h qa0, qa1;
  {
    const float* s0 = Q + (long long)qrow * DHEAD + hl * 8;
    #pragma unroll
    for (int e = 0; e < 8; ++e) {
      qa0[e]     = (_Float16)(s0[e]      * 0.125f);
      qa0[e + 8] = (_Float16)(s0[e + 16] * 0.125f);
    }
    const float* s1 = s0 + 32;
    #pragma unroll
    for (int e = 0; e < 8; ++e) {
      qa1[e]     = (_Float16)(s1[e]      * 0.125f);
      qa1[e + 8] = (_Float16)(s1[e + 16] * 0.125f);
    }
  }

  // ---- "ones in column 0" B fragment: row-sum rides the PV WMMA ----
  v16h bsum;
  {
    const _Float16 one = (_Float16)((l16 == 0) ? 1.0f : 0.0f);
    #pragma unroll
    for (int e = 0; e < 16; ++e) bsum[e] = one;
  }

  // ---- per-row valid key range (loop-invariant) ----
  int lo[8], hi[8];
  #pragma unroll
  for (int r = 0; r < 8; ++r) {
    const int i = q0 + wave * WROWS + r + hl * 8;
    lo[r] = (i - WIN > 0) ? (i - WIN) : 0;
    hi[r] = (i + WIN < L_SEQ - 1) ? (i + WIN) : (L_SEQ - 1);
  }

  float mrow[8];
  v8f   oacc[5];                    // 4 dim tiles + softmax-denominator tile
  #pragma unroll
  for (int r = 0; r < 8; ++r) mrow[r] = NEG_BIG;
  #pragma unroll
  for (int u = 0; u < 5; ++u)
    #pragma unroll
    for (int r = 0; r < 8; ++r) oacc[u][r] = 0.f;

  const int kbase0 = q0 - WIN;
  const int qw     = q0 + wave * WROWS;
  const int cmin   = wave >> 2;     // waves 0-3: chunks 0..4; waves 4-7: chunks 1..5
  const int cmax   = cmin + 4;

  for (int c = 0; c < NCHUNK; ++c) {
    __syncthreads();   // previous iteration's LDS reads done before restage

    // ---- stage K (row-major) and V (transposed, pair-packed b32 stores) ----
    // 256 threads: key pair = tid>>3 (0..31), dim segment = (tid&7)*8
    {
      const int pr   = tid >> 3;
      const int dseg = (tid & 7) * 8;
      const int kg0  = kbase0 + c * CHUNK + 2 * pr;
      _Float16 tk0[8], tk1[8], tv0[8], tv1[8];
      if (kg0 >= 0 && kg0 < L_SEQ) {
        const float* ks = K + (long long)kg0 * DHEAD + dseg;
        const float* vs = V + (long long)kg0 * DHEAD + dseg;
        #pragma unroll
        for (int e = 0; e < 8; ++e) { tk0[e] = (_Float16)ks[e]; tv0[e] = (_Float16)vs[e]; }
        if (kg0 + CHUNK < L_SEQ) {
          __builtin_prefetch(ks + CHUNK * DHEAD, 0, 1);
          __builtin_prefetch(vs + CHUNK * DHEAD, 0, 1);
        }
      } else {
        #pragma unroll
        for (int e = 0; e < 8; ++e) { tk0[e] = (_Float16)0.f; tv0[e] = (_Float16)0.f; }
      }
      if (kg0 + 1 >= 0 && kg0 + 1 < L_SEQ) {
        const float* ks = K + (long long)(kg0 + 1) * DHEAD + dseg;
        const float* vs = V + (long long)(kg0 + 1) * DHEAD + dseg;
        #pragma unroll
        for (int e = 0; e < 8; ++e) { tk1[e] = (_Float16)ks[e]; tv1[e] = (_Float16)vs[e]; }
      } else {
        #pragma unroll
        for (int e = 0; e < 8; ++e) { tk1[e] = (_Float16)0.f; tv1[e] = (_Float16)0.f; }
      }
      #pragma unroll
      for (int e = 0; e < 8; ++e) {
        sK[2 * pr    ][dseg + e] = tk0[e];
        sK[2 * pr + 1][dseg + e] = tk1[e];
      }
      #pragma unroll
      for (int e = 0; e < 8; ++e) {
        v2h p; p[0] = tv0[e]; p[1] = tv1[e];
        *(v2h*)&sVt[dseg + e][2 * pr] = p;    // packed b32 transpose store
      }
    }
    __syncthreads();

    const bool active = (c >= cmin) && (c <= cmax);   // wave-uniform

    v8f acc[4];
    if (active) {
      // ---- S = Q K^T : four 16-key tiles ----
      #pragma unroll
      for (int t = 0; t < 4; ++t) {
        const _Float16* krow = &sK[t * 16 + l16][0];
        v16h b0 = cat16(*(const v8h*)(krow + hl * 8),
                        *(const v8h*)(krow + hl * 8 + 16));        // dims 0..31
        v16h b1 = cat16(*(const v8h*)(krow + 32 + hl * 8),
                        *(const v8h*)(krow + 32 + hl * 8 + 16));   // dims 32..63
        v8f a;
        #pragma unroll
        for (int r = 0; r < 8; ++r) a[r] = 0.f;
        a = __builtin_amdgcn_wmma_f32_16x16x32_f16(false, qa0, false, b0, (short)0, a, false, false);
        a = __builtin_amdgcn_wmma_f32_16x16x32_f16(false, qa1, false, b1, (short)0, a, false, false);
        acc[t] = a;
      }

      // ---- mask only when this chunk can violate band/bounds ----
      const int  kb = kbase0 + c * CHUNK;
      const bool needMask = !((kb >= qw - 113) && (kb <= qw + 65) &&
                              (kb >= 0) && (kb + CHUNK <= L_SEQ));
      float tmax[8];
      #pragma unroll
      for (int r = 0; r < 8; ++r) tmax[r] = NEG_BIG;
      if (needMask) {
        #pragma unroll
        for (int t = 0; t < 4; ++t) {
          const int j = kb + t * 16 + l16;
          #pragma unroll
          for (int r = 0; r < 8; ++r) {
            const float sv = (j >= lo[r] && j <= hi[r]) ? acc[t][r] : NEG_BIG;
            acc[t][r] = sv;
            tmax[r] = fmaxf(tmax[r], sv);
          }
        }
      } else {
        #pragma unroll
        for (int t = 0; t < 4; ++t)
          #pragma unroll
          for (int r = 0; r < 8; ++r) tmax[r] = fmaxf(tmax[r], acc[t][r]);
      }
      // one butterfly per 64 keys: row-max across the 16 lanes holding each row
      #pragma unroll
      for (int msk = 1; msk < 16; msk <<= 1)
        #pragma unroll
        for (int r = 0; r < 8; ++r)
          tmax[r] = fmaxf(tmax[r], __shfl_xor(tmax[r], msk, 32));

      // ---- online softmax update (row-sum handled by the WMMA ones-column) ----
      float scl[8];
      #pragma unroll
      for (int r = 0; r < 8; ++r) {
        const float mn = fmaxf(mrow[r], tmax[r]);
        scl[r]  = __expf(mrow[r] - mn);
        mrow[r] = mn;
      }
      #pragma unroll
      for (int t = 0; t < 4; ++t)
        #pragma unroll
        for (int r = 0; r < 8; ++r) {
          const float p = __expf(acc[t][r] - mrow[r]);
          sP[wave][r + hl * 8][t * 16 + l16] = (_Float16)p;
        }
      #pragma unroll
      for (int u = 0; u < 5; ++u)
        #pragma unroll
        for (int r = 0; r < 8; ++r) oacc[u][r] *= scl[r];
    }

    __syncthreads();   // sP visible across lanes; all waves participate

    if (active) {
      // ---- O += P V (A = P 16x64 via two frags; B = V^T slabs + ones column) ----
      const _Float16* prow = &sP[wave][l16][0];
      v16h pf0 = cat16(*(const v8h*)(prow + hl * 8),
                       *(const v8h*)(prow + hl * 8 + 16));        // keys 0..31
      v16h pf1 = cat16(*(const v8h*)(prow + 32 + hl * 8),
                       *(const v8h*)(prow + 32 + hl * 8 + 16));   // keys 32..63
      #pragma unroll
      for (int t = 0; t < 4; ++t) {
        const _Float16* vrow = &sVt[t * 16 + l16][0];
        v16h bv0 = cat16(*(const v8h*)(vrow + hl * 8),
                         *(const v8h*)(vrow + hl * 8 + 16));
        v16h bv1 = cat16(*(const v8h*)(vrow + 32 + hl * 8),
                         *(const v8h*)(vrow + 32 + hl * 8 + 16));
        oacc[t] = __builtin_amdgcn_wmma_f32_16x16x32_f16(false, pf0, false, bv0,
                                                         (short)0, oacc[t], false, false);
        oacc[t] = __builtin_amdgcn_wmma_f32_16x16x32_f16(false, pf1, false, bv1,
                                                         (short)0, oacc[t], false, false);
      }
      oacc[4] = __builtin_amdgcn_wmma_f32_16x16x32_f16(false, pf0, false, bsum,
                                                       (short)0, oacc[4], false, false);
      oacc[4] = __builtin_amdgcn_wmma_f32_16x16x32_f16(false, pf1, false, bsum,
                                                       (short)0, oacc[4], false, false);
    }
  }

  // ---- normalize: denominator lives in column 0 of oacc[4]; broadcast & store ----
  #pragma unroll
  for (int r = 0; r < 8; ++r) {
    const float ssum = __shfl(oacc[4][r], hl * 16, 32);   // lane 0 / lane 16 hold col 0
    const float inv  = (ssum > 0.f) ? (1.0f / ssum) : 0.f;
    const long long row = (long long)(q0 + wave * WROWS + r + hl * 8);
    float* dst = O + row * DHEAD;
    #pragma unroll
    for (int t = 0; t < 4; ++t) dst[t * 16 + l16] = oacc[t][r] * inv;
  }
}

extern "C" void kernel_launch(void* const* d_in, const int* in_sizes, int n_in,
                              void* d_out, int out_size, void* d_ws, size_t ws_size,
                              hipStream_t stream) {
  const float* q = (const float*)d_in[0];
  const float* k = (const float*)d_in[1];
  const float* v = (const float*)d_in[2];
  float* out = (float*)d_out;

  const int bh = in_sizes[0] / (L_SEQ * DHEAD);   // B*H = 24
  dim3 grid(L_SEQ / QTILE, bh);
  dim3 block(256);
  lsa_fwd<<<grid, block, 0, stream>>>(q, k, v, out);
}
